// MyGraphNetwork0000_39685497815927
// MI455X (gfx1250) — matrix operand order repaired and checked by reading, not compiled, not run on hardware
//
#include <hip/hip_runtime.h>
#include <hip/hip_fp16.h>

typedef __attribute__((ext_vector_type(16))) _Float16 v16h;
typedef __attribute__((ext_vector_type(8)))  float    v8f;

#define FDIM 64

// ---------- helpers ----------
__device__ __forceinline__ unsigned fkey(float f) {          // monotone float->uint map
  unsigned u = __float_as_uint(f);
  return (u & 0x80000000u) ? ~u : (u | 0x80000000u);
}
__device__ __forceinline__ float funkey(unsigned k) {
  return (k & 0x80000000u) ? __uint_as_float(k & 0x7FFFFFFFu) : __uint_as_float(~k);
}
__device__ __forceinline__ float lrelu(float f) { return f > 0.f ? f : 0.2f * f; }

// ---------- init ----------
__global__ void zero_f(float* p, long long n) {
  long long i = (long long)blockIdx.x * blockDim.x + threadIdx.x;
  if (i < n) p[i] = 0.f;
}
__global__ void init_mkey(unsigned* p, int n) {
  int i = blockIdx.x * blockDim.x + threadIdx.x;
  if (i < n) p[i] = fkey(-__builtin_inff());
}

// ---------- degrees ----------
__global__ void edge_count(const long long* __restrict__ dst, int E, float* __restrict__ cnt) {
  int e = blockIdx.x * blockDim.x + threadIdx.x;
  if (e < E) atomicAdd(&cnt[(int)dst[e]], 1.0f);
}
__global__ void node_scalars(const float* __restrict__ cnt, float* __restrict__ dinv,
                             float* __restrict__ rcnt, int n) {
  int i = blockIdx.x * blockDim.x + threadIdx.x;
  if (i >= n) return;
  float c = cnt[i];
  dinv[i] = rsqrtf(c + 1.0f);          // GCN degree includes self-loop
  rcnt[i] = 1.0f / fmaxf(c, 1.0f);     // SAGE mean divisor
}

// ---------- WMMA GEMM: D[N,64] = A[N,64] @ W[64,64] (+bias)(+addend)(relu) ----------
// W is staged once per block into LDS, transposed + pre-converted to f16, so each
// B fragment is one contiguous 32B LDS read. One wave computes a 16x64 row strip:
// 4 col tiles x 2 k-steps = 8 v_wmma_f32_16x16x32_f16 per wave.
template <bool HAS_BIAS, bool HAS_ADD, bool RELU>
__global__ __launch_bounds__(256) void gemm_n64(
    const float* __restrict__ A, const float* __restrict__ W,
    const float* __restrict__ bias, const float* __restrict__ addend,
    float* __restrict__ D, int nRows) {
  __shared__ __align__(32) _Float16 Wt[FDIM * FDIM];  // Wt[n*64+k] = W[k*64+n], 8 KB

  // Cooperative stage: coalesced float4 row reads, scattered f16 transpose stores.
  {
    const int t  = threadIdx.x;
    const int k  = t >> 2;            // W row (K index)
    const int n0 = (t & 3) << 4;      // 16-column chunk
    const float4* wp = (const float4*)(W + (size_t)k * FDIM + n0);
    float4 q0 = wp[0], q1 = wp[1], q2 = wp[2], q3 = wp[3];
    Wt[(n0 +  0) * FDIM + k] = (_Float16)q0.x;
    Wt[(n0 +  1) * FDIM + k] = (_Float16)q0.y;
    Wt[(n0 +  2) * FDIM + k] = (_Float16)q0.z;
    Wt[(n0 +  3) * FDIM + k] = (_Float16)q0.w;
    Wt[(n0 +  4) * FDIM + k] = (_Float16)q1.x;
    Wt[(n0 +  5) * FDIM + k] = (_Float16)q1.y;
    Wt[(n0 +  6) * FDIM + k] = (_Float16)q1.z;
    Wt[(n0 +  7) * FDIM + k] = (_Float16)q1.w;
    Wt[(n0 +  8) * FDIM + k] = (_Float16)q2.x;
    Wt[(n0 +  9) * FDIM + k] = (_Float16)q2.y;
    Wt[(n0 + 10) * FDIM + k] = (_Float16)q2.z;
    Wt[(n0 + 11) * FDIM + k] = (_Float16)q2.w;
    Wt[(n0 + 12) * FDIM + k] = (_Float16)q3.x;
    Wt[(n0 + 13) * FDIM + k] = (_Float16)q3.y;
    Wt[(n0 + 14) * FDIM + k] = (_Float16)q3.z;
    Wt[(n0 + 15) * FDIM + k] = (_Float16)q3.w;
  }
  __syncthreads();   // before any wave-uniform exit

  const int lane = threadIdx.x & 31;
  const int wave = threadIdx.x >> 5;
  const int m    = lane & 15;
  const int hi   = lane >> 4;                       // half-wave select
  const int rowBase = (blockIdx.x * 8 + wave) * 16; // 8 row-tiles per block
  if (rowBase >= nRows) return;                     // wave-uniform exit

  const int row = rowBase + m;
  const float* arow = A + (size_t)(row < nRows ? row : 0) * FDIM;

  // A fragments for both k-steps: lane(hi) holds K = kb + hi*16 + {0..15}
  v16h af[2];
#pragma unroll
  for (int ks = 0; ks < 2; ++ks) {
    const float4* a4 = (const float4*)(arow + ks * 32 + hi * 16); // 16B aligned
    float4 p0 = a4[0], p1 = a4[1], p2 = a4[2], p3 = a4[3];
    af[ks][0]  = (_Float16)p0.x; af[ks][1]  = (_Float16)p0.y;
    af[ks][2]  = (_Float16)p0.z; af[ks][3]  = (_Float16)p0.w;
    af[ks][4]  = (_Float16)p1.x; af[ks][5]  = (_Float16)p1.y;
    af[ks][6]  = (_Float16)p1.z; af[ks][7]  = (_Float16)p1.w;
    af[ks][8]  = (_Float16)p2.x; af[ks][9]  = (_Float16)p2.y;
    af[ks][10] = (_Float16)p2.z; af[ks][11] = (_Float16)p2.w;
    af[ks][12] = (_Float16)p3.x; af[ks][13] = (_Float16)p3.y;
    af[ks][14] = (_Float16)p3.z; af[ks][15] = (_Float16)p3.w;
  }

  v8f acc[4];
#pragma unroll
  for (int t = 0; t < 4; ++t) {
    const int col = t * 16 + m;
    acc[t] = {};
#pragma unroll
    for (int ks = 0; ks < 2; ++ks) {
      // B fragment: lane holds column col for K = ks*32 + hi*16 + j
      // -> contiguous 16 halves in Wt: single 32B LDS read (2 x ds_load_b128)
      const v16h bf = *(const v16h*)(Wt + col * FDIM + ks * 32 + hi * 16);
      acc[t] = __builtin_amdgcn_wmma_f32_16x16x32_f16(
          false, af[ks], false, bf, (short)0, acc[t], false, false);
    }
  }

  // Epilogue: reg r -> row = rowBase + r + hi*8, col = t*16 + m
  const bool full = (rowBase + 16 <= nRows);
  if (full) {
#pragma unroll
    for (int t = 0; t < 4; ++t) {
      const int col = t * 16 + m;
      const float b = HAS_BIAS ? bias[col] : 0.f;
#pragma unroll
      for (int r = 0; r < 8; ++r) {
        const size_t o = (size_t)(rowBase + r + hi * 8) * FDIM + col;
        float v = acc[t][r] + b;
        if (HAS_ADD) v += addend[o];
        if (RELU)    v = fmaxf(v, 0.f);
        D[o] = v;
      }
    }
  } else {
#pragma unroll
    for (int t = 0; t < 4; ++t) {
      const int col = t * 16 + m;
      const float b = HAS_BIAS ? bias[col] : 0.f;
#pragma unroll
      for (int r = 0; r < 8; ++r) {
        const int orow = rowBase + r + hi * 8;
        if (orow < nRows) {
          const size_t o = (size_t)orow * FDIM + col;
          float v = acc[t][r] + b;
          if (HAS_ADD) v += addend[o];
          if (RELU)    v = fmaxf(v, 0.f);
          D[o] = v;
        }
      }
    }
  }
}

// ---------- GAT per-node attention scores ----------
__global__ void gat_scores(const float* __restrict__ hg, const float* __restrict__ a_src,
                           const float* __restrict__ a_dst, float* __restrict__ es,
                           float* __restrict__ ed, int n) {
  int i = blockIdx.x * blockDim.x + threadIdx.x;
  if (i >= n) return;
  const float* r = hg + (size_t)i * FDIM;
  float s = 0.f, d = 0.f;
#pragma unroll 8
  for (int k = 0; k < FDIM; ++k) { float v = r[k]; s += v * a_src[k]; d += v * a_dst[k]; }
  es[i] = s; ed[i] = d;
}

// ---------- edge aggregations (warp-per-edge, 2 features/lane) ----------
__global__ __launch_bounds__(256) void edge_agg_sum(
    const long long* __restrict__ src, const long long* __restrict__ dst,
    const float* __restrict__ feat, float* __restrict__ acc, int E) {
  int e = blockIdx.x * 8 + (threadIdx.x >> 5);
  if (e >= E) return;
  int lane = threadIdx.x & 31;
  int s = (int)src[e], d = (int)dst[e];
  const float* fs = feat + (size_t)s * FDIM;
  float* ad = acc + (size_t)d * FDIM;
  atomicAdd(&ad[lane],      fs[lane]);
  atomicAdd(&ad[lane + 32], fs[lane + 32]);
}

__global__ __launch_bounds__(256) void edge_agg_gcn(
    const long long* __restrict__ src, const long long* __restrict__ dst,
    const float* __restrict__ h, const float* __restrict__ dinv,
    float* __restrict__ acc, int E) {
  int e = blockIdx.x * 8 + (threadIdx.x >> 5);
  if (e >= E) return;
  int lane = threadIdx.x & 31;
  int s = (int)src[e], d = (int)dst[e];
  float nrm = dinv[s] * dinv[d];
  const float* hs = h + (size_t)s * FDIM;
  float* ad = acc + (size_t)d * FDIM;
  atomicAdd(&ad[lane],      hs[lane]      * nrm);
  atomicAdd(&ad[lane + 32], hs[lane + 32] * nrm);
}

__global__ void gat_edge_max(const long long* __restrict__ src, const long long* __restrict__ dst,
                             const float* __restrict__ es, const float* __restrict__ ed,
                             unsigned* __restrict__ mkey, int E) {
  int e = blockIdx.x * blockDim.x + threadIdx.x;
  if (e >= E) return;
  int s = (int)src[e], d = (int)dst[e];
  atomicMax(&mkey[d], fkey(lrelu(es[s] + ed[d])));
}

__global__ void gat_node_max(const unsigned* __restrict__ mkey, const float* __restrict__ es,
                             const float* __restrict__ ed, float* __restrict__ mflt, int n) {
  int i = blockIdx.x * blockDim.x + threadIdx.x;
  if (i >= n) return;
  mflt[i] = fmaxf(funkey(mkey[i]), lrelu(es[i] + ed[i]));   // include self-loop edge
}

__global__ __launch_bounds__(256) void gat_edge_acc(
    const long long* __restrict__ src, const long long* __restrict__ dst,
    const float* __restrict__ es, const float* __restrict__ ed,
    const float* __restrict__ mflt, const float* __restrict__ hg,
    float* __restrict__ Z, float* __restrict__ num, int E) {
  int e = blockIdx.x * 8 + (threadIdx.x >> 5);
  if (e >= E) return;
  int lane = threadIdx.x & 31;
  int s = (int)src[e], d = (int)dst[e];
  float x = __expf(lrelu(es[s] + ed[d]) - mflt[d]);
  if (lane == 0) atomicAdd(&Z[d], x);
  const float* hs = hg + (size_t)s * FDIM;
  float* nd = num + (size_t)d * FDIM;
  atomicAdd(&nd[lane],      x * hs[lane]);
  atomicAdd(&nd[lane + 32], x * hs[lane + 32]);
}

// ---------- elementwise ----------
__global__ void prep_gin_mean(const float* __restrict__ x, float* __restrict__ sumx,
                              float* __restrict__ gin_in, const float* __restrict__ rcnt,
                              long long n) {
  long long i = (long long)blockIdx.x * blockDim.x + threadIdx.x;
  if (i >= n) return;
  float sv = sumx[i];
  gin_in[i] = x[i] + sv;            // GIN input (eps=0)
  sumx[i] = sv * rcnt[i >> 6];      // becomes SAGE mean (in place)
}

__global__ void gcn_final(float* __restrict__ acc, const float* __restrict__ h,
                          const float* __restrict__ dinv, const float* __restrict__ b,
                          long long n) {
  long long i = (long long)blockIdx.x * blockDim.x + threadIdx.x;
  if (i >= n) return;
  int node = (int)(i >> 6), k = (int)(i & 63);
  float dv = dinv[node];
  acc[i] = fmaxf(acc[i] + h[i] * dv * dv + b[k], 0.f);       // + self-loop, + bias, relu
}

__global__ void gat_final(float* __restrict__ num, const float* __restrict__ hg,
                          const float* __restrict__ es, const float* __restrict__ ed,
                          const float* __restrict__ mflt, const float* __restrict__ Z,
                          const float* __restrict__ b, long long n) {
  long long i = (long long)blockIdx.x * blockDim.x + threadIdx.x;
  if (i >= n) return;
  int node = (int)(i >> 6), k = (int)(i & 63);
  float e_self = lrelu(es[node] + ed[node]);
  float xs = __expf(e_self - mflt[node]);
  float Zf = Z[node] + xs;
  num[i] = fmaxf((num[i] + xs * hg[i]) / Zf + b[k], 0.f);
}

// ---------- output head: sigmoid(concat · W_out + b_out), F_OUT = 1 ----------
__global__ void head(const float* __restrict__ gcn, const float* __restrict__ gat,
                     const float* __restrict__ sage, const float* __restrict__ gin,
                     const float* __restrict__ W_out, const float* __restrict__ b_out,
                     float* __restrict__ out, int n) {
  int i = blockIdx.x * blockDim.x + threadIdx.x;
  if (i >= n) return;
  size_t off = (size_t)i * FDIM;
  float s = b_out[0];
#pragma unroll 8
  for (int k = 0; k < FDIM; ++k) {
    s += gcn[off + k]  * W_out[k]
       + gat[off + k]  * W_out[64 + k]
       + sage[off + k] * W_out[128 + k]
       + gin[off + k]  * W_out[192 + k];
  }
  out[i] = 1.f / (1.f + __expf(-s));
}

// ---------- launch ----------
extern "C" void kernel_launch(void* const* d_in, const int* in_sizes, int n_in,
                              void* d_out, int out_size, void* d_ws, size_t ws_size,
                              hipStream_t stream) {
  const float*     x        = (const float*)d_in[0];
  const long long* ei       = (const long long*)d_in[1];
  const float*     W_gcn    = (const float*)d_in[2];
  const float*     b_gcn    = (const float*)d_in[3];
  const float*     W_gat    = (const float*)d_in[4];
  const float*     a_src    = (const float*)d_in[5];
  const float*     a_dst    = (const float*)d_in[6];
  const float*     b_gat    = (const float*)d_in[7];
  const float*     W_sage_l = (const float*)d_in[8];
  const float*     b_sage_l = (const float*)d_in[9];
  const float*     W_sage_r = (const float*)d_in[10];
  const float*     W_gin    = (const float*)d_in[11];
  const float*     b_gin    = (const float*)d_in[12];
  const float*     W_out    = (const float*)d_in[13];
  const float*     b_out    = (const float*)d_in[14];

  const int N = in_sizes[0] / FDIM;
  const int E = in_sizes[1] / 2;
  const long long* srcI = ei;
  const long long* dstI = ei + E;

  const size_t NE = (size_t)N * FDIM;
  float* base   = (float*)d_ws;
  float* B_h    = base;            // x @ W_gcn
  float* B_hg   = base + 1 * NE;   // x @ W_gat
  float* B_gcn  = base + 2 * NE;   // GCN accumulator -> relu(gcn)
  float* B_gat  = base + 3 * NE;   // GAT numerator   -> relu(gat)
  float* B_sumx = base + 4 * NE;   // sum_x -> mean (in place)
  float* B_t1   = base + 5 * NE;   // gin_in
  float* B_t2   = base + 6 * NE;   // relu(gin)
  float* B_t3   = base + 7 * NE;   // xr -> relu(sage) (fused addend, in place)
  float* cnt  = base + 8 * NE;
  float* dinv = cnt + N;  float* rcnt = dinv + N;
  float* es   = rcnt + N; float* ed   = es + N;
  float* mflt = ed + N;   float* Z    = mflt + N;
  unsigned* mkey = (unsigned*)(Z + N);

  auto cdiv = [](long long a, long long b) { return (unsigned)((a + b - 1) / b); };

  // 0) re-init accumulators every call (graph-replay safe)
  zero_f<<<cdiv(3 * (long long)NE, 256), 256, 0, stream>>>(B_gcn, 3 * (long long)NE); // gcn, gat, sumx
  zero_f<<<cdiv(7LL * N, 256), 256, 0, stream>>>(cnt, 7LL * N);                       // cnt..Z
  init_mkey<<<cdiv(N, 256), 256, 0, stream>>>(mkey, N);

  // 1) degrees
  edge_count<<<cdiv(E, 256), 256, 0, stream>>>(dstI, E, cnt);
  node_scalars<<<cdiv(N, 256), 256, 0, stream>>>(cnt, dinv, rcnt, N);

  // 2) dense projections via WMMA (one wave per 16x64 strip; 8 strips per block)
  const unsigned gb = cdiv(N, 128);
  gemm_n64<false, false, false><<<gb, 256, 0, stream>>>(x, W_gcn,    nullptr, nullptr, B_h,  N);
  gemm_n64<false, false, false><<<gb, 256, 0, stream>>>(x, W_gat,    nullptr, nullptr, B_hg, N);
  gemm_n64<false, false, false><<<gb, 256, 0, stream>>>(x, W_sage_r, nullptr, nullptr, B_t3, N);

  // 3) attention scalars
  gat_scores<<<cdiv(N, 256), 256, 0, stream>>>(B_hg, a_src, a_dst, es, ed, N);

  // 4) edge passes (L2-resident scatter/gather)
  edge_agg_sum<<<cdiv(E, 8), 256, 0, stream>>>(srcI, dstI, x, B_sumx, E);
  edge_agg_gcn<<<cdiv(E, 8), 256, 0, stream>>>(srcI, dstI, B_h, dinv, B_gcn, E);
  gat_edge_max<<<cdiv(E, 256), 256, 0, stream>>>(srcI, dstI, es, ed, mkey, E);
  gat_node_max<<<cdiv(N, 256), 256, 0, stream>>>(mkey, es, ed, mflt, N);
  gat_edge_acc<<<cdiv(E, 8), 256, 0, stream>>>(srcI, dstI, es, ed, mflt, B_hg, Z, B_gat, E);

  // 5) SAGE mean + GIN input, then WMMA GEMMs with fused bias/addend/relu
  prep_gin_mean<<<cdiv((long long)NE, 256), 256, 0, stream>>>(x, B_sumx, B_t1, rcnt, (long long)NE);
  gemm_n64<true, true, true ><<<gb, 256, 0, stream>>>(B_sumx, W_sage_l, b_sage_l, B_t3, B_t3, N); // relu(mean@Wl + xr + b)
  gemm_n64<true, false, true><<<gb, 256, 0, stream>>>(B_t1,   W_gin,    b_gin,    nullptr, B_t2, N); // relu((x+sum)@Wg + b)

  // 6) finalize GCN/GAT (self-loops + bias + relu)
  gcn_final<<<cdiv((long long)NE, 256), 256, 0, stream>>>(B_gcn, B_h, dinv, b_gcn, (long long)NE);
  gat_final<<<cdiv((long long)NE, 256), 256, 0, stream>>>(B_gat, B_hg, es, ed, mflt, Z, b_gat, (long long)NE);

  // 7) head
  head<<<cdiv(N, 256), 256, 0, stream>>>(B_gcn, B_gat, B_t3, B_t2, W_out, b_out, (float*)d_out, N);
}
